// SparseAutoencoder_4209067950442
// MI455X (gfx1250) — matrix operand order repaired and checked by reading, not compiled, not run on hardware
//
#include <hip/hip_runtime.h>
#include <hip/hip_bf16.h>
#include <stdint.h>

#define B_SZ 8192
#define D_SZ 768
#define H_SZ 16384
#define NNZ_CAP 64

typedef __attribute__((ext_vector_type(16))) __bf16        v16bf;
typedef __attribute__((ext_vector_type(8)))  float         v8f;
typedef __attribute__((ext_vector_type(4)))  unsigned int  v4u;
typedef __attribute__((ext_vector_type(8)))  unsigned int  v8u;

__device__ __forceinline__ unsigned short f32_to_bf16_rne(float f) {
    unsigned u = __builtin_bit_cast(unsigned, f);
    unsigned r = u + 0x7FFFu + ((u >> 16) & 1u);
    return (unsigned short)(r >> 16);
}

// ---------------- prep: xc = x - b_pre -> bf16 ----------------
__global__ void __launch_bounds__(256) k_prep_x(const float* __restrict__ x,
                                                const float* __restrict__ b_pre,
                                                unsigned short* __restrict__ xcbf) {
    size_t i = (size_t)blockIdx.x * 256 + threadIdx.x;
    int d = (int)(i % D_SZ);
    float v = __builtin_nontemporal_load(&x[i]) - b_pre[d];
    xcbf[i] = f32_to_bf16_rne(v);
}

// ---------------- prep: W_enc -> bf16 ----------------
__global__ void __launch_bounds__(256) k_prep_w(const float* __restrict__ w,
                                                unsigned short* __restrict__ wbf) {
    size_t i = (size_t)blockIdx.x * 256 + threadIdx.x;
    wbf[i] = f32_to_bf16_rne(__builtin_nontemporal_load(&w[i]));
}

// ---------------- prep: W_dec (D,H) -> W_decT (H,D) ----------------
__global__ void __launch_bounds__(256) k_transpose_wdec(const float* __restrict__ Wdec,
                                                        float* __restrict__ WdT) {
    __shared__ float tile[32][33];
    const int h0 = blockIdx.x * 32;
    const int d0 = blockIdx.y * 32;
    const int tx = threadIdx.x & 31;
    const int ty = threadIdx.x >> 5;   // 0..7
#pragma unroll
    for (int k = 0; k < 4; ++k) {
        int d = d0 + ty + 8 * k;
        tile[ty + 8 * k][tx] = __builtin_nontemporal_load(&Wdec[(size_t)d * H_SZ + h0 + tx]);
    }
    __syncthreads();
#pragma unroll
    for (int k = 0; k < 4; ++k) {
        int h = h0 + ty + 8 * k;
        WdT[(size_t)h * D_SZ + d0 + tx] = tile[tx][ty + 8 * k];
    }
}

// ---------------- WMMA fragment loader (layouts per CDNA5 ISA 7.12.2) ----------------
__device__ __forceinline__ v16bf load_frag2(const unsigned short* p_lo,
                                            const unsigned short* p_hi) {
    v4u lo = *reinterpret_cast<const v4u*>(p_lo);
    v4u hi = *reinterpret_cast<const v4u*>(p_hi);
    v8u w = {lo[0], lo[1], lo[2], lo[3], hi[0], hi[1], hi[2], hi[3]};
    return __builtin_bit_cast(v16bf, w);
}

// ---------------- encoder GEMM: Z(B,H) = xc(B,D) * W_enc(H,D)^T, bf16 WMMA ----------------
// Wave tile: 32(M) x 64(N), software-pipelined K loop (K step = 32).
__global__ void __launch_bounds__(256) k_gemm_enc(const unsigned short* __restrict__ A,
                                                  const unsigned short* __restrict__ Wb,
                                                  float* __restrict__ Z) {
    const int lane  = threadIdx.x & 31;
    const int wave  = threadIdx.x >> 5;        // 0..7
    const int m     = lane & 15;
    const int half  = lane >> 4;
    const int cA    = half * 8;                // A k-chunk base within tile
    const int kB    = half * 8;                // B k-chunk base within tile
    const int col0  = blockIdx.x * 64;         // H tile
    const int row0  = blockIdx.y * 256 + wave * 32;  // B tile (per wave: 32 rows)

    const v8f z8 = {0.f, 0.f, 0.f, 0.f, 0.f, 0.f, 0.f, 0.f};
    v8f acc[2][4];
#pragma unroll
    for (int i = 0; i < 2; ++i)
#pragma unroll
        for (int j = 0; j < 4; ++j) acc[i][j] = z8;

    // Per-lane base pointers (already offset by the lane's k-chunk origin).
    const unsigned short* aR0 = A + (size_t)(row0 + m) * D_SZ + cA;
    const unsigned short* aR1 = A + (size_t)(row0 + 16 + m) * D_SZ + cA;
    const unsigned short* bR[4];
#pragma unroll
    for (int j = 0; j < 4; ++j)
        bR[j] = Wb + (size_t)(col0 + j * 16 + m) * D_SZ + kB;   // n = lane&15

    // ---- prologue: load k=0 fragments ----
    v16bf a0 = load_frag2(aR0, aR0 + 16);
    v16bf a1 = load_frag2(aR1, aR1 + 16);
    v16bf b[4];
#pragma unroll
    for (int j = 0; j < 4; ++j)
        b[j] = load_frag2(bR[j], bR[j] + 16);

    // ---- pipelined main loop: issue loads for k0, compute with k0-32 ----
#pragma unroll 2
    for (int k0 = 32; k0 < D_SZ; k0 += 32) {
        v16bf na0 = load_frag2(aR0 + k0, aR0 + k0 + 16);
        v16bf na1 = load_frag2(aR1 + k0, aR1 + k0 + 16);
        v16bf nb[4];
#pragma unroll
        for (int j = 0; j < 4; ++j)
            nb[j] = load_frag2(bR[j] + k0, bR[j] + k0 + 16);

#pragma unroll
        for (int j = 0; j < 4; ++j) {
            acc[0][j] = __builtin_amdgcn_wmma_f32_16x16x32_bf16(
                false, a0, false, b[j], (short)0, acc[0][j], false, false);
            acc[1][j] = __builtin_amdgcn_wmma_f32_16x16x32_bf16(
                false, a1, false, b[j], (short)0, acc[1][j], false, false);
        }

        a0 = na0; a1 = na1;
#pragma unroll
        for (int j = 0; j < 4; ++j) b[j] = nb[j];
    }

    // ---- epilogue: compute with the last fragments ----
#pragma unroll
    for (int j = 0; j < 4; ++j) {
        acc[0][j] = __builtin_amdgcn_wmma_f32_16x16x32_bf16(
            false, a0, false, b[j], (short)0, acc[0][j], false, false);
        acc[1][j] = __builtin_amdgcn_wmma_f32_16x16x32_bf16(
            false, a1, false, b[j], (short)0, acc[1][j], false, false);
    }

    // C/D layout: VGPR r -> row (r + 8*half), col = lane&15.
    // Non-temporal: Z is a 512MB stream, not re-read until the next kernel.
#pragma unroll
    for (int i = 0; i < 2; ++i) {
#pragma unroll
        for (int j = 0; j < 4; ++j) {
            size_t base = (size_t)(row0 + i * 16 + half * 8) * H_SZ + col0 + j * 16 + m;
#pragma unroll
            for (int r = 0; r < 8; ++r)
                __builtin_nontemporal_store(acc[i][j][r], &Z[base + (size_t)r * H_SZ]);
        }
    }
}

// ---------------- per-row top-k radix select + mask + compact ----------------
__global__ void __launch_bounds__(256) k_topk(float* __restrict__ Z,
                                              const int* __restrict__ kptr,
                                              int* __restrict__ nnz_cnt,
                                              int* __restrict__ nnz_idx,
                                              float* __restrict__ nnz_val) {
    __shared__ unsigned hist[256];
    __shared__ unsigned s_pfx;
    __shared__ int s_rem;
    __shared__ int s_cnt;
    __shared__ int   s_idx[NNZ_CAP];
    __shared__ float s_valv[NNZ_CAP];

    const int row = blockIdx.x;
    const int tid = threadIdx.x;
    float* zrow = Z + (size_t)row * H_SZ;
    const int kk = kptr[0];

    if (tid == 0) { s_rem = kk; s_pfx = 0u; s_cnt = 0; }
    __syncthreads();

    // 4-pass byte radix select of k-th largest key (monotonic f32->u32 map)
    for (int shift = 24; shift >= 0; shift -= 8) {
        hist[tid] = 0u;
        __syncthreads();
        const unsigned maskHi = (shift == 24) ? 0u : (0xFFFFFFFFu << (shift + 8));
        const unsigned pfx = s_pfx;
        for (int i = tid; i < H_SZ; i += 256) {
            unsigned u = __builtin_bit_cast(unsigned, zrow[i]);
            unsigned key = (u & 0x80000000u) ? ~u : (u | 0x80000000u);
            if ((key & maskHi) == pfx)
                atomicAdd(&hist[(key >> shift) & 255u], 1u);
        }
        __syncthreads();
        if (tid == 0) {
            int rem = s_rem;
            unsigned cum = 0;
            int chosen = 0;
            for (int b = 255; b >= 0; --b) {
                unsigned c = hist[b];
                if (cum + c >= (unsigned)rem) { chosen = b; break; }
                cum += c;
            }
            s_rem = rem - (int)cum;
            s_pfx = pfx | ((unsigned)chosen << shift);
        }
        __syncthreads();
    }

    const unsigned T = s_pfx;  // key of k-th largest; keep key >= T  (== z >= thresh)
    for (int i = tid; i < H_SZ; i += 256) {
        float f = zrow[i];
        unsigned u = __builtin_bit_cast(unsigned, f);
        unsigned key = (u & 0x80000000u) ? ~u : (u | 0x80000000u);
        bool keep = key >= T;
        zrow[i] = keep ? f : 0.0f;
        if (keep) {
            int p = atomicAdd(&s_cnt, 1);
            if (p < NNZ_CAP) { s_idx[p] = i; s_valv[p] = f; }
        }
    }
    __syncthreads();

    if (tid == 0) {
        int n = s_cnt; if (n > NNZ_CAP) n = NNZ_CAP;
        // deterministic order: insertion sort by index
        for (int a = 1; a < n; ++a) {
            int ia = s_idx[a]; float va = s_valv[a];
            int b = a - 1;
            while (b >= 0 && s_idx[b] > ia) {
                s_idx[b + 1] = s_idx[b]; s_valv[b + 1] = s_valv[b]; --b;
            }
            s_idx[b + 1] = ia; s_valv[b + 1] = va;
        }
        s_cnt = n;
        nnz_cnt[row] = n;
    }
    __syncthreads();

    if (tid < s_cnt) {
        nnz_idx[(size_t)row * NNZ_CAP + tid] = s_idx[tid];
        nnz_val[(size_t)row * NNZ_CAP + tid] = s_valv[tid];
    }
}

// ---------------- sparse decode: dec(B,D) = enc_sparse * W_decT + b_pre ----------------
__global__ void __launch_bounds__(256) k_decode(const float* __restrict__ WdT,
                                                const float* __restrict__ b_pre,
                                                const int* __restrict__ nnz_cnt,
                                                const int* __restrict__ nnz_idx,
                                                const float* __restrict__ nnz_val,
                                                float* __restrict__ Dec) {
    const int row = blockIdx.x;
    const int tid = threadIdx.x;
    float acc0 = b_pre[tid];
    float acc1 = b_pre[tid + 256];
    float acc2 = b_pre[tid + 512];
    const int n = nnz_cnt[row];
    for (int j = 0; j < n; ++j) {
        float v = nnz_val[(size_t)row * NNZ_CAP + j];
        int   h = nnz_idx[(size_t)row * NNZ_CAP + j];
        const float* w = WdT + (size_t)h * D_SZ;
        acc0 += v * w[tid];
        acc1 += v * w[tid + 256];
        acc2 += v * w[tid + 512];
    }
    float* out = Dec + (size_t)row * D_SZ;
    out[tid]       = acc0;
    out[tid + 256] = acc1;
    out[tid + 512] = acc2;
}

extern "C" void kernel_launch(void* const* d_in, const int* in_sizes, int n_in,
                              void* d_out, int out_size, void* d_ws, size_t ws_size,
                              hipStream_t stream) {
    const float* x        = (const float*)d_in[0];
    const float* W_enc    = (const float*)d_in[1];
    const float* W_dec    = (const float*)d_in[2];
    const float* b_pre    = (const float*)d_in[3];
    const int*   k_sparse = (const int*)d_in[4];

    float* enc = (float*)d_out;                      // B*H
    float* dec = enc + (size_t)B_SZ * H_SZ;          // B*D

    char* ws = (char*)d_ws;
    size_t o = 0;
    auto align256 = [](size_t v) { return (v + 255) & ~(size_t)255; };

    unsigned short* xcbf = (unsigned short*)(ws + o); o = align256(o + (size_t)B_SZ * D_SZ * 2);
    unsigned short* webf = (unsigned short*)(ws + o); o = align256(o + (size_t)H_SZ * D_SZ * 2);
    float*          wdT  = (float*)(ws + o);          o = align256(o + (size_t)H_SZ * D_SZ * 4);
    int*   nnz_cnt = (int*)(ws + o);   o = align256(o + (size_t)B_SZ * 4);
    int*   nnz_idx = (int*)(ws + o);   o = align256(o + (size_t)B_SZ * NNZ_CAP * 4);
    float* nnz_val = (float*)(ws + o); o = align256(o + (size_t)B_SZ * NNZ_CAP * 4);
    (void)ws_size; (void)in_sizes; (void)n_in; (void)out_size;

    k_prep_x<<<(B_SZ * D_SZ) / 256, 256, 0, stream>>>(x, b_pre, xcbf);
    k_prep_w<<<(H_SZ * D_SZ) / 256, 256, 0, stream>>>(W_enc, webf);

    dim3 gT(H_SZ / 32, D_SZ / 32);
    k_transpose_wdec<<<gT, 256, 0, stream>>>(W_dec, wdT);

    dim3 gG(H_SZ / 64, B_SZ / 256);
    k_gemm_enc<<<gG, 256, 0, stream>>>(xcbf, webf, enc);

    k_topk<<<B_SZ, 256, 0, stream>>>(enc, k_sparse, nnz_cnt, nnz_idx, nnz_val);
    k_decode<<<B_SZ, 256, 0, stream>>>(wdT, b_pre, nnz_cnt, nnz_idx, nnz_val, dec);
}
